// STPCell_33706903339377
// MI455X (gfx1250) — matrix operand order repaired and checked by reading, not compiled, not run on hardware
//
#include <hip/hip_runtime.h>
#include <hip/hip_bf16.h>

#define H_    1024
#define IN_   512
#define B_    32
#define ZMIN_ 0.001f
#define ZRNG_ 0.099f   /* Z_MAX - Z_MIN */
#define EH_   0.5f

typedef __attribute__((ext_vector_type(16))) _Float16 v16h;
typedef __attribute__((ext_vector_type(8)))  float    v8f;
typedef __attribute__((ext_vector_type(4)))  float    f32x4;

__device__ __forceinline__ float sigf(float v) {
  return __builtin_amdgcn_rcpf(1.0f + __expf(-v));
}

// One element of the fused X_new/U_new/W chain, FMA-refactored.
__device__ __forceinline__ float stp_elem(float Xv, float Uv, float zx, float zu,
                                          float uc, float wv, float hk, float hbj) {
  float Xn = fmaf(Xv, (1.0f - zx) - Uv * hbj, zx);
  float Un = fmaf(Uv, (1.0f - zu) - uc * hbj, uc * (zu + hbj));
  Un = fminf(fmaxf(Un, uc), 1.0f);            // clip(U_new, Ucap, 1.0)
  return (wv * Un) * (Xn * hk);               // w * U_new * X_new * h[b,k]
}

// ---------------------------------------------------------------------------
// px = p (1024x512) @ x (512x32)  via v_wmma_f32_16x16x32_f16.
// One wave per 16x16 output tile; 128 tiles; K looped in steps of 32.
// Layouts per CDNA5 ISA 7.12.2 (16-bit A 16x32, B 32x16, f32 C 16x16).
// ---------------------------------------------------------------------------
__global__ __launch_bounds__(128) void px_gemm_wmma(const float* __restrict__ P,
                                                    const float* __restrict__ Xin,
                                                    float* __restrict__ px) {
  const int lane = threadIdx.x & 31;
  const int tile = blockIdx.x * 4 + (threadIdx.x >> 5);  // 0..127
  const int tm   = tile >> 1;                            // 0..63 (M/16)
  const int tn   = tile & 1;                             // 0..1  (N/16)
  const int r16  = lane & 15;
  const int kg   = lane >> 4;                            // half-wave K group
  const int m    = tm * 16 + r16;                        // A row (both halves map M=0..15)
  const int n    = tn * 16 + r16;                        // B column

  v8f c = {};
  for (int kb = 0; kb < IN_; kb += 32) {
    v16h a, bm;
#pragma unroll
    for (int i = 0; i < 8; ++i) {
      // A 16x32 f16: VGPR i<4 -> K = 8*kg + 2i(+1); i>=4 -> K = 16 + 8*kg + 2(i-4)(+1)
      const int kA = 2 * i + (i < 4 ? 0 : 8) + 8 * kg;
      a[2 * i]     = (_Float16)P[(size_t)m * IN_ + kb + kA];
      a[2 * i + 1] = (_Float16)P[(size_t)m * IN_ + kb + kA + 1];
      // B 32x16 f16: same K mapping as A (i<4 -> K = 8*kg + 2i(+1); i>=4 -> K = 16 + 8*kg + 2(i-4)(+1))
      const int kB = 2 * i + (i < 4 ? 0 : 8) + 8 * kg;
      bm[2 * i]     = (_Float16)Xin[(size_t)(kb + kB) * B_ + n];
      bm[2 * i + 1] = (_Float16)Xin[(size_t)(kb + kB + 1) * B_ + n];
    }
    c = __builtin_amdgcn_wmma_f32_16x16x32_f16(false, a, false, bm,
                                               (short)0, c, false, false);
  }
  // C f32 16x16: VGPR r, lanes 0-15 -> M=r, lanes 16-31 -> M=8+r; N = lane&15
#pragma unroll
  for (int r = 0; r < 8; ++r) {
    const int M = tm * 16 + r + 8 * kg;
    const int N = tn * 16 + r16;
    px[(size_t)M * B_ + N] = c[r];
  }
}

// ---------------------------------------------------------------------------
// Main streaming kernel. Grid (8, 128), block 256 (8 waves).
// Wave = one j row x 4 batches: params & sigmoids amortized 4x.
// X/U streamed once with NON-TEMPORAL b128 loads (keep 192MB L2 for the
// 16MB of reused params), butterfly-reduced over 32 lanes.
// ---------------------------------------------------------------------------
__global__ __launch_bounds__(256) void stp_main(
    const float* __restrict__ h,  const float* __restrict__ X,
    const float* __restrict__ U,  const float* __restrict__ cx,
    const float* __restrict__ cu, const float* __restrict__ cUc,
    const float* __restrict__ ch, const float* __restrict__ w,
    const float* __restrict__ px, const float* __restrict__ bias,
    float* __restrict__ out) {
  const int lane = threadIdx.x & 31;
  const int wave = threadIdx.x >> 5;
  const int b0   = blockIdx.x * 4;            // batch group: 0,4,...,28
  const int j    = blockIdx.y * 8 + wave;     // 0..1023

  const size_t prow = (size_t)j * H_;
  const f32x4* cx4 = (const f32x4*)(cx  + prow);
  const f32x4* cu4 = (const f32x4*)(cu  + prow);
  const f32x4* cU4 = (const f32x4*)(cUc + prow);
  const f32x4* w4  = (const f32x4*)(w   + prow);

  float hbj[4];
  const f32x4* X4[4];
  const f32x4* U4[4];
  const f32x4* h4[4];
#pragma unroll
  for (int q = 0; q < 4; ++q) {
    const int b = b0 + q;
    hbj[q] = h[(size_t)b * H_ + j];
    const size_t row = ((size_t)b * H_ + j) * H_;
    X4[q] = (const f32x4*)(X + row);
    U4[q] = (const f32x4*)(U + row);
    h4[q] = (const f32x4*)(h + (size_t)b * H_);
  }

  float acc[4] = {0.f, 0.f, 0.f, 0.f};
#pragma unroll 2
  for (int i = 0; i < H_ / 128; ++i) {        // 8 iterations, k = 128 per iter
    const int idx = i * 32 + lane;
    const f32x4 cxv = cx4[idx], cuv = cu4[idx], cUv = cU4[idx], wv = w4[idx];
    f32x4 zx, zu, uc;
    zx.x = ZMIN_ + ZRNG_ * sigf(cxv.x);  zx.y = ZMIN_ + ZRNG_ * sigf(cxv.y);
    zx.z = ZMIN_ + ZRNG_ * sigf(cxv.z);  zx.w = ZMIN_ + ZRNG_ * sigf(cxv.w);
    zu.x = ZMIN_ + ZRNG_ * sigf(cuv.x);  zu.y = ZMIN_ + ZRNG_ * sigf(cuv.y);
    zu.z = ZMIN_ + ZRNG_ * sigf(cuv.z);  zu.w = ZMIN_ + ZRNG_ * sigf(cuv.w);
    uc.x = 0.9f * sigf(cUv.x);           uc.y = 0.9f * sigf(cUv.y);
    uc.z = 0.9f * sigf(cUv.z);           uc.w = 0.9f * sigf(cUv.w);
#pragma unroll
    for (int q = 0; q < 4; ++q) {
      // zero-reuse streams: non-temporal so they bypass L2 residency
      const f32x4 xv = __builtin_nontemporal_load(&X4[q][idx]);
      const f32x4 uv = __builtin_nontemporal_load(&U4[q][idx]);
      const f32x4 hv = h4[q][idx];
      const float hb = hbj[q];
      float s;
      s  = stp_elem(xv.x, uv.x, zx.x, zu.x, uc.x, wv.x, hv.x, hb);
      s += stp_elem(xv.y, uv.y, zx.y, zu.y, uc.y, wv.y, hv.y, hb);
      s += stp_elem(xv.z, uv.z, zx.z, zu.z, uc.z, wv.z, hv.z, hb);
      s += stp_elem(xv.w, uv.w, zx.w, zu.w, uc.w, wv.w, hv.w, hb);
      acc[q] += s;
    }
  }

  // wave32 butterfly reduction -> every lane holds the full row sum
#pragma unroll
  for (int q = 0; q < 4; ++q)
    for (int off = 16; off; off >>= 1)
      acc[q] += __shfl_xor(acc[q], off);

  if (lane == 0) {
    const float zh = EH_ * sigf(ch[j]);
#pragma unroll
    for (int q = 0; q < 4; ++q) {
      const int b = b0 + q;
      const float pre = acc[q] + px[(size_t)j * B_ + b] + bias[j];
      out[(size_t)b * H_ + j] = (1.0f - zh) * hbj[q] + zh * sigf(pre);
    }
  }
}

extern "C" void kernel_launch(void* const* d_in, const int* in_sizes, int n_in,
                              void* d_out, int out_size, void* d_ws, size_t ws_size,
                              hipStream_t stream) {
  const float* x    = (const float*)d_in[0];   // (512, 32)
  const float* h    = (const float*)d_in[1];   // (32, 1024)
  const float* X    = (const float*)d_in[2];   // (32, 1024, 1024)
  const float* U    = (const float*)d_in[3];   // (32, 1024, 1024)
  const float* c_x  = (const float*)d_in[4];   // (1024, 1024)
  const float* c_u  = (const float*)d_in[5];
  const float* c_U  = (const float*)d_in[6];
  const float* c_h  = (const float*)d_in[7];   // (1024, 1)
  const float* w    = (const float*)d_in[8];   // (1024, 1024)
  const float* p    = (const float*)d_in[9];   // (1024, 512)
  const float* bias = (const float*)d_in[10];  // (1024, 1)
  float* out = (float*)d_out;                  // (32, 1024)
  float* px  = (float*)d_ws;                   // 1024*32 floats = 128 KB scratch

  // 1) px = p @ x via WMMA (128 tiles, 4 waves/block)
  px_gemm_wmma<<<dim3(32, 1), 128, 0, stream>>>(p, x, px);
  // 2) fused streaming update + reduction + output
  stp_main<<<dim3(8, 128), 256, 0, stream>>>(h, X, U, c_x, c_u, c_U, c_h, w,
                                             px, bias, out);
}
